// ImprovedAttention_26207890440645
// MI455X (gfx1250) — compile-verified, matrix operand order checked by reading, not run on hardware
//
#include <hip/hip_runtime.h>
#include <math.h>

// ---- CDNA5 WMMA vector types ----
typedef __bf16 v16bf __attribute__((ext_vector_type(16)));
typedef __bf16 v8bf  __attribute__((ext_vector_type(8)));
typedef float  v8f   __attribute__((ext_vector_type(8)));
typedef float  v4f   __attribute__((ext_vector_type(4)));

#define B_   64
#define S_   2048
#define H_   512
#define K_   1024   // 2*H, reduction dim of the big GEMM
#define MT   64     // S rows per workgroup (A tile height)
#define NCH  64     // N chunk width staged per iteration
#define AST  1032   // LDS A row stride in halves (1024 + 8 pad -> conflict-free frags)
#define BST  1032   // LDS B row stride in halves

// fp32 -> bf16 round-to-nearest-even
__device__ __forceinline__ unsigned short f2bf(float x) {
    unsigned int u = __float_as_uint(x);
    u += 0x7FFFu + ((u >> 16) & 1u);
    return (unsigned short)(u >> 16);
}

union Frag { v16bf v; v8bf h[2]; };

// ---------------- Kernel A: dec_proj = decoder_hidden @ Wd (64x512x512) ----------------
__global__ void __launch_bounds__(256) dec_proj_kernel(
    const float* __restrict__ dh, const float* __restrict__ Wd, float* __restrict__ dp)
{
    __shared__ float s[H_];
    const int b = blockIdx.x, tid = threadIdx.x;
    s[tid]       = dh[(size_t)b * H_ + tid];
    s[tid + 256] = dh[(size_t)b * H_ + tid + 256];
    __syncthreads();
    float a0 = 0.f, a1 = 0.f;
    for (int k = 0; k < H_; ++k) {
        const float t = s[k];
        a0 += t * Wd[(size_t)k * H_ + tid];
        a1 += t * Wd[(size_t)k * H_ + tid + 256];
    }
    dp[(size_t)b * H_ + tid]       = a0;
    dp[(size_t)b * H_ + tid + 256] = a1;
}

// ---------------- Kernel B: fused enc_proj GEMM (bf16 WMMA) + tanh + v-dot ----------------
// grid = (S/MT, B), block = 256 (8 wave32), dynamic LDS ~268 KB (<= 320 KB/WGP)
// Each wave: one A fragment feeds TWO B tiles (3 ds_load_b128 per WMMA).
__global__ void __launch_bounds__(256) bahdanau_scores_kernel(
    const float* __restrict__ enc, const float* __restrict__ We,
    const float* __restrict__ dp,  const float* __restrict__ v,
    float* __restrict__ scores)
{
    extern __shared__ char smem[];
    unsigned short* sA  = (unsigned short*)smem;        // MT x AST bf16 (full-K A tile)
    unsigned short* sB  = sA + MT * AST;                // NCH x BST bf16 (We chunk, [n][k])
    float*          sDV = (float*)(sB + NCH * BST);     // interleaved (dec_proj, v): 2*H
    float*          sS  = sDV + 2 * H_;                 // MT score accumulators

    const int tid = threadIdx.x;
    const int b   = blockIdx.y;
    const int s0  = blockIdx.x * MT;

    if (tid < MT) sS[tid] = 0.0f;
    // Stage (dec_proj[b,:], v[:]) interleaved -> one ds_load_b64 in the epilogue
    sDV[2 * tid]             = dp[(size_t)b * H_ + tid];
    sDV[2 * tid + 1]         = v[tid];
    sDV[2 * (tid + 256)]     = dp[(size_t)b * H_ + tid + 256];
    sDV[2 * (tid + 256) + 1] = v[tid + 256];

    // Stage A tile: MT x K fp32 -> bf16 in LDS (full K staged once, non-temporal reads)
    const float* Abase = enc + ((size_t)b * S_ + s0) * (size_t)K_;
    for (int e = tid * 4; e < MT * K_; e += 256 * 4) {
        const int m = e >> 10;            // /1024
        const int k = e & (K_ - 1);
        const v4f f = __builtin_nontemporal_load((const v4f*)(Abase + (size_t)m * K_ + k));
        unsigned int* d = (unsigned int*)(sA + m * AST + k);
        d[0] = (unsigned)f2bf(f.x) | ((unsigned)f2bf(f.y) << 16);
        d[1] = (unsigned)f2bf(f.z) | ((unsigned)f2bf(f.w) << 16);
    }
    __syncthreads();

    const int lane  = tid & 31;
    const int wave  = tid >> 5;
    const int l16   = lane & 15;
    const int half  = lane >> 4;
    const int msub  = wave & 3;    // 4 m-subtiles of 16 rows
    const int nhalf = wave >> 2;   // 2 n-groups of 32 cols per chunk

    // Fragment bases per ISA 7.12.2 (16-bit A 16x32; B mirrors with N across lanes)
    const unsigned short* aBase  = sA + (msub * 16 + l16) * AST + half * 8;
    const unsigned short* bBase0 = sB + (nhalf * 32 + l16) * BST + half * 8;
    const unsigned short* bBase1 = bBase0 + 16 * BST;

    for (int nb = 0; nb < H_; nb += NCH) {
        // Stage We[:, nb:nb+NCH] transposed -> sB[n][k] (bf16), float4-coalesced reads
        {
            const int nloc = (tid & 15) * 4;           // 0..60
            for (int k = tid >> 4; k < K_; k += 16) {
                const v4f f = *(const v4f*)(We + (size_t)k * H_ + nb + nloc);
                sB[(nloc + 0) * BST + k] = f2bf(f.x);
                sB[(nloc + 1) * BST + k] = f2bf(f.y);
                sB[(nloc + 2) * BST + k] = f2bf(f.z);
                sB[(nloc + 3) * BST + k] = f2bf(f.w);
            }
        }
        __syncthreads();

        v8f c0 = {}, c1 = {};
        #pragma unroll 4
        for (int k0 = 0; k0 < K_; k0 += 32) {
            Frag fa, fb0, fb1;
            fa.h[0]  = *(const v8bf*)(aBase  + k0);
            fa.h[1]  = *(const v8bf*)(aBase  + k0 + 16);
            fb0.h[0] = *(const v8bf*)(bBase0 + k0);
            fb0.h[1] = *(const v8bf*)(bBase0 + k0 + 16);
            fb1.h[0] = *(const v8bf*)(bBase1 + k0);
            fb1.h[1] = *(const v8bf*)(bBase1 + k0 + 16);
            c0 = __builtin_amdgcn_wmma_f32_16x16x32_bf16(
                     false, fa.v, false, fb0.v, (short)0, c0, false, false);
            c1 = __builtin_amdgcn_wmma_f32_16x16x32_bf16(
                     false, fa.v, false, fb1.v, (short)0, c1, false, false);
        }

        // Epilogue: energy = tanh(c + dec_proj[n]); score_m += v[n]*energy (both tiles)
        const int ng0 = nb + nhalf * 32 + l16;
        const float2 dv0 = *(const float2*)(sDV + 2 * ng0);
        const float2 dv1 = *(const float2*)(sDV + 2 * (ng0 + 16));
        #pragma unroll
        for (int r = 0; r < 8; ++r) {
            float e = tanhf(c0[r] + dv0.x) * dv0.y
                    + tanhf(c1[r] + dv1.x) * dv1.y;
            e += __shfl_xor(e, 1);
            e += __shfl_xor(e, 2);
            e += __shfl_xor(e, 4);
            e += __shfl_xor(e, 8);
            // C layout: VGPR r holds row r (lanes 0-15) and row r+8 (lanes 16-31)
            if (l16 == 0) atomicAdd(&sS[msub * 16 + half * 8 + r], e);
        }
        __syncthreads();
    }

    if (tid < MT) scores[(size_t)b * S_ + s0 + tid] = sS[tid];
}

// ---------------- Kernel C: masked softmax over S per batch; zero context ----------------
__global__ void __launch_bounds__(256) softmax_kernel(
    const float* __restrict__ scores, const int* __restrict__ mask,
    float* __restrict__ attn, float* __restrict__ ctx)
{
    __shared__ float red[256];
    const int b = blockIdx.x, tid = threadIdx.x;
    float vals[8];
    float mx = -3.4e38f;
    #pragma unroll
    for (int i = 0; i < 8; ++i) {
        const int idx = tid + i * 256;
        float x = scores[(size_t)b * S_ + idx];
        if (mask[(size_t)b * S_ + idx] == 0) x = -1e10f;
        vals[i] = x;
        mx = fmaxf(mx, x);
    }
    red[tid] = mx; __syncthreads();
    for (int s = 128; s > 0; s >>= 1) {
        if (tid < s) red[tid] = fmaxf(red[tid], red[tid + s]);
        __syncthreads();
    }
    mx = red[0]; __syncthreads();

    float sum = 0.f;
    #pragma unroll
    for (int i = 0; i < 8; ++i) { vals[i] = __expf(vals[i] - mx); sum += vals[i]; }
    red[tid] = sum; __syncthreads();
    for (int s = 128; s > 0; s >>= 1) {
        if (tid < s) red[tid] += red[tid + s];
        __syncthreads();
    }
    const float inv = 1.0f / red[0];
    #pragma unroll
    for (int i = 0; i < 8; ++i)
        attn[(size_t)b * S_ + tid + i * 256] = vals[i] * inv;
    #pragma unroll
    for (int i = 0; i < 4; ++i)
        ctx[(size_t)b * K_ + tid + i * 256] = 0.0f;
}

// ---------------- Kernel D: context = attn @ encoder_outputs ----------------
// grid = (S/64 chunks, B); non-temporal float4 streaming; global f32 atomics across chunks
__global__ void __launch_bounds__(256) context_kernel(
    const float* __restrict__ enc, const float* __restrict__ attn,
    float* __restrict__ ctx)
{
    __shared__ float w[64];
    const int b = blockIdx.y, s0 = blockIdx.x * 64, tid = threadIdx.x;
    if (tid < 64) w[tid] = attn[(size_t)b * S_ + s0 + tid];
    __syncthreads();
    const int d = tid * 4;
    float ax = 0.f, ay = 0.f, az = 0.f, aw = 0.f;
    const float* base = enc + ((size_t)b * S_ + s0) * (size_t)K_ + d;
    for (int s = 0; s < 64; ++s) {
        const float ws = w[s];
        const v4f e4 = __builtin_nontemporal_load((const v4f*)(base + (size_t)s * K_));
        ax += ws * e4.x; ay += ws * e4.y; az += ws * e4.z; aw += ws * e4.w;
    }
    atomicAdd(&ctx[(size_t)b * K_ + d + 0], ax);
    atomicAdd(&ctx[(size_t)b * K_ + d + 1], ay);
    atomicAdd(&ctx[(size_t)b * K_ + d + 2], az);
    atomicAdd(&ctx[(size_t)b * K_ + d + 3], aw);
}

extern "C" void kernel_launch(void* const* d_in, const int* in_sizes, int n_in,
                              void* d_out, int out_size, void* d_ws, size_t ws_size,
                              hipStream_t stream)
{
    const float* dec  = (const float*)d_in[0];   // (64, 512)
    const float* enc  = (const float*)d_in[1];   // (64, 2048, 1024)
    const int*   mask = (const int*)  d_in[2];   // (64, 2048)
    const float* Wd   = (const float*)d_in[3];   // (512, 512)
    const float* We   = (const float*)d_in[4];   // (1024, 512)
    const float* v    = (const float*)d_in[5];   // (512,)

    float* out    = (float*)d_out;
    float* ctx    = out;                          // (64, 1024)
    float* attn   = out + (size_t)B_ * K_;        // (64, 2048)
    float* dp     = (float*)d_ws;                 // (64, 512)
    float* scores = dp + (size_t)B_ * H_;         // (64, 2048)

    dec_proj_kernel<<<B_, 256, 0, stream>>>(dec, Wd, dp);

    const size_t smem = (size_t)(MT * AST + NCH * BST) * sizeof(unsigned short)
                      + (size_t)(2 * H_ + MT) * sizeof(float);   // ~268.6 KB
    bahdanau_scores_kernel<<<dim3(S_ / MT, B_), 256, smem, stream>>>(enc, We, dp, v, scores);

    softmax_kernel<<<B_, 256, 0, stream>>>(scores, mask, attn, ctx);

    context_kernel<<<dim3(S_ / 64, B_), 256, 0, stream>>>(enc, attn, ctx);
}
